// GDF_MIL_74234214744525
// MI455X (gfx1250) — compile-verified, matrix operand order checked by reading, not run on hardware
//
#include <hip/hip_runtime.h>

typedef __attribute__((ext_vector_type(16))) __bf16        v16bf;
typedef __attribute__((ext_vector_type(8)))  float         v8f;
typedef __attribute__((ext_vector_type(8)))  unsigned int  v8u;

#define DEVINL __device__ __forceinline__
#define CS_ASYNC 1   // use global_load_async_to_lds_b128 for panel staging

constexpr int   NSAMP = 100000;
constexpr int   DIN   = 512;
constexpr int   DH    = 256;
constexpr int   DO    = 128;
constexpr int   KC    = 1024;
constexpr int   TK    = 16;
constexpr float NEGS  = 0.01f;

DEVINL float lrelu(float v){ return v > 0.f ? v : NEGS * v; }
DEVINL float sigm (float v){ return 1.f / (1.f + __expf(-v)); }
DEVINL unsigned short f2bfu(float f){ __bf16 h = (__bf16)f; return __builtin_bit_cast(unsigned short, h); }
DEVINL float bfu2f(unsigned short u){ return (float)__builtin_bit_cast(__bf16, u); }

// k-index of element e (0..15) of a 16x32 bf16 A/B fragment for this lane
// (ISA 7.12.2: VGPR0..3 lanes0-15 K=0..7, lanes16-31 K=8..15; VGPR4..7 +16)
DEVINL int kmap(int e, int lane){
  int vg = e >> 1, j = e & 1, kb = (lane >> 4) * 8;
  return (vg < 4) ? (kb + vg * 2 + j) : (16 + kb + (vg - 4) * 2 + j);
}

DEVINL v8f wmma_bf16(v16bf a, v16bf b, v8f c){
  return __builtin_amdgcn_wmma_f32_16x16x32_bf16(false, a, false, b, (short)0, c, false, false);
}

// Load one 16x32 fragment for this lane. Element (row,k) lives at
// p[rowoff + k*kstride] (elements, not bytes). UNIT=1 means kstride==1 so
// each (k,k+1) pair is a single aligned dword load (merged to b128 by LLVM).
template<int F32, int UNIT>
DEVINL v16bf load_frag(const void* __restrict__ p, long rowoff, long kstride, int k0, int lane){
  v8u u;
#pragma unroll
  for (int pr = 0; pr < 8; pr++){
    int kk = k0 + kmap(2 * pr, lane);           // even element of the pair
    if constexpr (UNIT){
      if constexpr (F32){
        const float* f = (const float*)p + rowoff + kk;
        unsigned short l = f2bfu(f[0]), h = f2bfu(f[1]);
        u[pr] = ((unsigned)h << 16) | l;
      } else {
        u[pr] = *(const unsigned*)((const unsigned short*)p + rowoff + kk);
      }
    } else {
      if constexpr (F32){
        const float* f = (const float*)p;
        unsigned short l = f2bfu(f[rowoff + (long)kk * kstride]);
        unsigned short h = f2bfu(f[rowoff + (long)(kk + 1) * kstride]);
        u[pr] = ((unsigned)h << 16) | l;
      } else {
        const unsigned short* q = (const unsigned short*)p;
        unsigned short l = q[rowoff + (long)kk * kstride];
        unsigned short h = q[rowoff + (long)(kk + 1) * kstride];
        u[pr] = ((unsigned)h << 16) | l;
      }
    }
  }
  return __builtin_bit_cast(v16bf, u);
}

// ---------------------------------------------------------------------------
// WMMA GEMM: C[M,N] = act(A[M,K] @ B[K,N] + bias). A element (m,k) at
// A[m*sAm + k*sAk]; B element (k,n) at B[n*sBn + k*sBk]. One 16x16 tile per
// wave; 8 waves (256 thr, wave32) per block; tiles of one block share tm.
// split>1: K range carved into `split` chunks (each a multiple of 32),
// partial tiles accumulated into Cf with global atomics (bias/act must be
// none in that mode; zero Cf first).
// ---------------------------------------------------------------------------
template<int AF32, int BF32, int AUNIT, int BUNIT>
__global__ void k_gemm_wmma(const void* __restrict__ A, long sAm, long sAk,
                            const void* __restrict__ B, long sBn, long sBk,
                            const float* __restrict__ bias,
                            float* __restrict__ Cf, unsigned short* __restrict__ Cbf,
                            int M, int Kd, int Ncol, int act,
                            int split, int atomicOut)
{
  int wave = blockIdx.x * (blockDim.x >> 5) + (threadIdx.x >> 5);
  int lane = threadIdx.x & 31;
  int tilesN = Ncol >> 4;
  int tiles  = (M >> 4) * tilesN;
  if (wave >= tiles * split) return;         // wave-uniform: EXEC stays all-1s
  int chunk = wave / tiles;
  int tile  = wave - chunk * tiles;
  int tm = (tile / tilesN) << 4;
  int tn = (tile % tilesN) << 4;
  int r15 = lane & 15;
  long aoff = (long)(tm + r15) * sAm;
  long boff = (long)(tn + r15) * sBn;

  int cs   = (((Kd + split - 1) / split + 31) & ~31);   // chunk length, %32==0
  int kBeg = chunk * cs;
  int kEnd = kBeg + cs < Kd ? kBeg + cs : Kd;

  v8f acc = {};
  for (int k0 = kBeg; k0 < kEnd; k0 += 32){
    // prefetch two K-steps ahead on streaming unit-stride operands
    int kn = k0 + 64;
    if (kn < kEnd){
      if constexpr (AUNIT)
        __builtin_prefetch(AF32 ? (const void*)((const float*)A + aoff + kn)
                                : (const void*)((const unsigned short*)A + aoff + kn), 0, 0);
      if constexpr (BUNIT)
        __builtin_prefetch(BF32 ? (const void*)((const float*)B + boff + kn)
                                : (const void*)((const unsigned short*)B + boff + kn), 0, 0);
    }
    v16bf a = load_frag<AF32, AUNIT>(A, aoff, sAk, k0, lane);
    v16bf b = load_frag<BF32, BUNIT>(B, boff, sBk, k0, lane);
    acc = wmma_bf16(a, b, acc);
  }
  int mb = (lane >> 4) * 8;
  if (atomicOut){
#pragma unroll
    for (int r = 0; r < 8; r++){
      int m = tm + mb + r, n = tn + r15;
      atomicAdd(&Cf[(long)m * Ncol + n], acc[r]);
    }
  } else {
#pragma unroll
    for (int r = 0; r < 8; r++){
      int m = tm + mb + r, n = tn + r15;
      float v = acc[r] + (bias ? bias[n] : 0.f);
      if (act == 1) v = lrelu(v); else if (act == 2) v = sigm(v);
      if (Cf)  Cf [(long)m * Ncol + n] = v;
      if (Cbf) Cbf[(long)m * Ncol + n] = f2bfu(v);
    }
  }
}

// transpose+convert W[K,N] (f32) -> Wt[N,K] (bf16) so B fragments are k-unit
__global__ void k_transpose_bf16(const float* __restrict__ W, unsigned short* __restrict__ Wt,
                                 int K, int N){
  int i = blockIdx.x * blockDim.x + threadIdx.x;
  if (i >= K * N) return;
  int k = i / N, n = i % N;
  Wt[(long)n * K + k] = f2bfu(W[i]);
}

// ---------------------------------------------------------------------------
// Fused cluster assignment: logits = (H @ cl_W + cl_b + gumbel) * 2,
// row softmax over K=1024, write P as bf16.  16 rows / block, 8 waves each
// computing a 16x128 slice (8 WMMA tiles).  H panel staged to LDS with the
// CDNA5 async global->LDS copy path.
// ---------------------------------------------------------------------------
__global__ void k_cluster_softmax(const unsigned short* __restrict__ Hbf,
                                  const unsigned short* __restrict__ clWt, // [KC][DH] bf16
                                  const float* __restrict__ clb,
                                  const float* __restrict__ gumbel,
                                  unsigned short* __restrict__ Pbf)
{
  __shared__ float sm[16 * KC];            // 64 KB logits
  __shared__ float red[256];
  __shared__ unsigned short Ha[16 * DH];   // 8 KB staged H panel
  int wave = threadIdx.x >> 5, lane = threadIdx.x & 31;
  int row0 = blockIdx.x * 16;
  int col0 = wave * 128;
  int r15  = lane & 15;

  { // stage 16x256 bf16 (8 KB) H panel: 512 x 16B chunks, 256 threads x 2
    const char* gbase = (const char*)(Hbf + (long)row0 * DH);
#if CS_ASYNC
    unsigned lbase = (unsigned)(unsigned long long)(&Ha[0]); // low 32 bits = LDS offset
#pragma unroll
    for (int r = 0; r < 2; r++){
      int t = threadIdx.x + r * 256;
      unsigned laddr = lbase + t * 16;
      const char* ga = gbase + t * 16;
      asm volatile("global_load_async_to_lds_b128 %0, %1, off"
                   :: "v"(laddr), "v"(ga) : "memory");
    }
    asm volatile("s_wait_asynccnt 0x0" ::: "memory");
#else
#pragma unroll
    for (int r = 0; r < 2; r++){
      int t = threadIdx.x + r * 256;
      ((uint4*)Ha)[t] = ((const uint4*)gbase)[t];
    }
#endif
  }
  __syncthreads();

  v8f acc[8] = {};
  for (int k0 = 0; k0 < DH; k0 += 32){
    v16bf a;
    {
      v8u u;
#pragma unroll
      for (int pr = 0; pr < 8; pr++){
        int kk = k0 + kmap(2 * pr, lane);
        u[pr] = *(const unsigned*)&Ha[r15 * DH + kk];
      }
      a = __builtin_bit_cast(v16bf, u);
    }
#pragma unroll
    for (int t = 0; t < 8; t++){
      v16bf b = load_frag<0, 1>(clWt, (long)(col0 + t * 16 + r15) * DH, 1, k0, lane);
      acc[t] = wmma_bf16(a, b, acc[t]);
    }
  }
  int mb = (lane >> 4) * 8;
#pragma unroll
  for (int t = 0; t < 8; t++)
#pragma unroll
    for (int r = 0; r < 8; r++){
      int m = mb + r, n = col0 + t * 16 + r15;
      sm[m * KC + n] = 2.0f * (acc[t][r] + clb[n] + gumbel[(long)(row0 + m) * KC + n]);
    }
  __syncthreads();

  for (int m = 0; m < 16; m++){
    float mx = -3.4e38f;
    for (int c = threadIdx.x; c < KC; c += 256) mx = fmaxf(mx, sm[m * KC + c]);
    red[threadIdx.x] = mx; __syncthreads();
    for (int s = 128; s > 0; s >>= 1){
      if (threadIdx.x < s) red[threadIdx.x] = fmaxf(red[threadIdx.x], red[threadIdx.x + s]);
      __syncthreads();
    }
    mx = red[0]; __syncthreads();
    float sum = 0.f;
    for (int c = threadIdx.x; c < KC; c += 256){
      float e = __expf(sm[m * KC + c] - mx);
      sm[m * KC + c] = e; sum += e;
    }
    red[threadIdx.x] = sum; __syncthreads();
    for (int s = 128; s > 0; s >>= 1){
      if (threadIdx.x < s) red[threadIdx.x] += red[threadIdx.x + s];
      __syncthreads();
    }
    float inv = 1.f / red[0]; __syncthreads();
    for (int c = threadIdx.x; c < KC; c += 256)
      Pbf[(long)(row0 + m) * KC + c] = f2bfu(sm[m * KC + c] * inv);
  }
}

// ---------------------------------------------------------------------------
// Small fp32 GEMM (K-scale graph ops): C = act(scale * A@B + bias + add)
// ---------------------------------------------------------------------------
__global__ void k_gemm_f32(const float* __restrict__ A, int sAm, int sAk,
                           const float* __restrict__ B, int sBk, int sBn,
                           const float* __restrict__ bias, const float* __restrict__ add,
                           float scale, float* __restrict__ C,
                           int M, int Kd, int Ncol, int act)
{
  int i = blockIdx.x * blockDim.x + threadIdx.x;
  if (i >= M * Ncol) return;
  int m = i / Ncol, n = i % Ncol;
  float s = 0.f;
  for (int k = 0; k < Kd; k++)
    s += A[(long)m * sAm + (long)k * sAk] * B[(long)k * sBk + (long)n * sBn];
  s = s * scale + (bias ? bias[n] : 0.f) + (add ? add[i] : 0.f);
  if (act == 1) s = lrelu(s); else if (act == 2) s = sigm(s);
  C[i] = s;
}

__global__ void k_zero(float* p, long n){
  long i = (long)blockIdx.x * blockDim.x + threadIdx.x;
  long st = (long)gridDim.x * blockDim.x;
  for (; i < n; i += st) p[i] = 0.f;
}

// top-16 + softmax over each row of the 1024x1024 edge-logit matrix
__global__ void k_topk(const float* __restrict__ L, float* __restrict__ wout, int* __restrict__ iout){
  int r = blockIdx.x * blockDim.x + threadIdx.x;
  if (r >= KC) return;
  float v[TK]; int id[TK];
#pragma unroll
  for (int t = 0; t < TK; t++){ v[t] = -3.4e38f; id[t] = 0; }
  for (int c = 0; c < KC; c++){
    float x = L[(long)r * KC + c];
    if (x > v[TK - 1]){
      int p = TK - 1;
      while (p > 0 && v[p - 1] < x){ v[p] = v[p - 1]; id[p] = id[p - 1]; p--; }
      v[p] = x; id[p] = c;
    }
  }
  float mx = v[0], s = 0.f;
#pragma unroll
  for (int t = 0; t < TK; t++){ v[t] = __expf(v[t] - mx); s += v[t]; }
#pragma unroll
  for (int t = 0; t < TK; t++){ wout[r * TK + t] = v[t] / s; iout[r * TK + t] = id[t]; }
}

// segment-sum of parts[src] into agg[dst] over the 16384 top-k edges
__global__ void k_sage_agg(const int* __restrict__ idx, const float* __restrict__ parts,
                           float* __restrict__ agg, float* __restrict__ cnt){
  int e = blockIdx.x;            // edge id; threadIdx.x = feature (256)
  int src = e >> 4, dst = idx[e];
  atomicAdd(&agg[(long)dst * DH + threadIdx.x], parts[(long)src * DH + threadIdx.x]);
  if (threadIdx.x == 0) atomicAdd(&cnt[dst], 1.0f);
}

__global__ void k_divcnt(float* agg, const float* cnt, int rows, int cols){
  int i = blockIdx.x * blockDim.x + threadIdx.x;
  if (i >= rows * cols) return;
  agg[i] /= fmaxf(cnt[i / cols], 1.0f);
}

__global__ void k_summed(const float* __restrict__ w, const int* __restrict__ idx,
                         const float* __restrict__ Xg, float* __restrict__ summed){
  int i = blockIdx.x * blockDim.x + threadIdx.x;
  if (i >= KC * DO) return;
  int k = i >> 7, d = i & 127;
  float s = 0.f;
#pragma unroll
  for (int e = 0; e < TK; e++)
    s += w[k * TK + e] * Xg[(long)idx[k * TK + e] * DO + d];
  summed[i] = s;
}

__global__ void k_ewise2(const float* a, const float* b, float* s, float* m, long n){
  long i = (long)blockIdx.x * blockDim.x + threadIdx.x;
  if (i >= n) return;
  s[i] = a[i] + b[i];
  m[i] = a[i] * b[i];
}

// gated combine + residual + LayerNorm (one block of 128 per cluster row)
__global__ void k_out_ln(const float* g, const float* sum_msg, const float* bi_msg,
                         const float* Xg, const float* lnw, const float* lnb, float* zn){
  int k = blockIdx.x, d = threadIdx.x;
  long i = (long)k * DO + d;
  __shared__ float sh[DO];
  float gv = g[i];
  float o = lrelu(gv * sum_msg[i] + (1.f - gv) * bi_msg[i]) + Xg[i];
  sh[d] = o; __syncthreads();
  for (int s = 64; s > 0; s >>= 1){ if (d < s) sh[d] += sh[d + s]; __syncthreads(); }
  float mu = sh[0] / (float)DO; __syncthreads();
  float dd = o - mu;
  sh[d] = dd * dd; __syncthreads();
  for (int s = 64; s > 0; s >>= 1){ if (d < s) sh[d] += sh[d + s]; __syncthreads(); }
  float var = sh[0] / (float)DO;
  zn[i] = dd * rsqrtf(var + 1e-5f) * lnw[d] + lnb[d];
}

// attention-pooled bag embedding over the K clusters
__global__ void k_bgnn(const float* __restrict__ gate, const float* __restrict__ zn,
                       float* __restrict__ bg){
  __shared__ float sh[256]; __shared__ float mx_s, sum_s;
  int t = threadIdx.x;
  float v = -3.4e38f;
  for (int k = t; k < KC; k += 256) v = fmaxf(v, gate[k]);
  sh[t] = v; __syncthreads();
  for (int s = 128; s > 0; s >>= 1){ if (t < s) sh[t] = fmaxf(sh[t], sh[t + s]); __syncthreads(); }
  if (t == 0) mx_s = sh[0]; __syncthreads();
  float s0 = 0.f;
  for (int k = t; k < KC; k += 256) s0 += __expf(gate[k] - mx_s);
  sh[t] = s0; __syncthreads();
  for (int s = 128; s > 0; s >>= 1){ if (t < s) sh[t] += sh[t + s]; __syncthreads(); }
  if (t == 0) sum_s = sh[0]; __syncthreads();
  if (t < DO){
    float a = 0.f;
    for (int k = 0; k < KC; k++) a += __expf(gate[k] - mx_s) * zn[(long)k * DO + t];
    bg[t] = a / sum_s;
  }
}

// per-sample gated attention score: sum_d lrelu(fA)*sigm(fB)*attC[d]
__global__ void k_scores(const unsigned short* sA, const unsigned short* sB,
                         const float* attC, float* scores){
  int n = blockIdx.x * blockDim.x + threadIdx.x;
  if (n >= NSAMP) return;
  float s = 0.f;
  for (int d = 0; d < DO; d++)
    s += bfu2f(sA[(long)n * DO + d]) * bfu2f(sB[(long)n * DO + d]) * attC[d];
  scores[n] = s;
}

__global__ void k_red_part(const float* s, int n, const float* gmax, float* out, int op){
  __shared__ float sh[256];
  int i = blockIdx.x * 256 + threadIdx.x;
  float gm = 0.f; if (op == 1) gm = gmax[0];
  float v = (op == 0) ? -3.4e38f : 0.f;
  if (i < n) v = (op == 0) ? s[i] : __expf(s[i] - gm);
  sh[threadIdx.x] = v; __syncthreads();
  for (int st = 128; st > 0; st >>= 1){
    if (threadIdx.x < st)
      sh[threadIdx.x] = (op == 0) ? fmaxf(sh[threadIdx.x], sh[threadIdx.x + st])
                                  : sh[threadIdx.x] + sh[threadIdx.x + st];
    __syncthreads();
  }
  if (threadIdx.x == 0) out[blockIdx.x] = sh[0];
}

__global__ void k_red_final(const float* part, int nb, float* out, int op){
  __shared__ float sh[256];
  int t = threadIdx.x;
  float v = (op == 0) ? -3.4e38f : 0.f;
  for (int i = t; i < nb; i += 256) v = (op == 0) ? fmaxf(v, part[i]) : v + part[i];
  sh[t] = v; __syncthreads();
  for (int st = 128; st > 0; st >>= 1){
    if (t < st) sh[t] = (op == 0) ? fmaxf(sh[t], sh[t + st]) : sh[t] + sh[t + st];
    __syncthreads();
  }
  if (t == 0) out[0] = sh[0];
}

// Af[d] = sum_n softmax(scores)_n * f[n,d]   (one block per feature d)
__global__ void k_att_pool(const float* sc, const float* gmax, const float* denom,
                           const unsigned short* fbf, float* Af){
  int d = blockIdx.x;
  __shared__ float sh[256];
  float acc = 0.f;
  for (int n = threadIdx.x; n < NSAMP; n += 256)
    acc += __expf(sc[n] - gmax[0]) * bfu2f(fbf[(long)n * DH + d]);
  sh[threadIdx.x] = acc; __syncthreads();
  for (int st = 128; st > 0; st >>= 1){
    if (threadIdx.x < st) sh[threadIdx.x] += sh[threadIdx.x + st];
    __syncthreads();
  }
  if (threadIdx.x == 0) Af[d] = sh[0] / denom[0];
}

// FeatureFusion + classifier head (single block of 256)
__global__ void k_head(const float* bg, const float* bb,
                       const float* fgW, const float* fgb,
                       const float* ftW, const float* ftb,
                       const float* c1W, const float* c1b,
                       const float* c2W, const float* c2b, float* out){
  __shared__ float comb[2 * DO], bvec[DO], h1[DH];
  int t = threadIdx.x;
  comb[t] = (t < DO) ? bg[t] : bb[t - DO];
  __syncthreads();
  if (t < DO){
    float sfg = fgb[t], sft = ftb[t];
    for (int k = 0; k < 2 * DO; k++){
      sfg += comb[k] * fgW[k * DO + t];
      sft += comb[k] * ftW[k * DO + t];
    }
    float fgv = sigm(sfg), ftv = lrelu(sft);
    bvec[t] = fgv * bg[t] + (1.f - fgv) * ftv;
  }
  __syncthreads();
  {
    float s = c1b[t];
    for (int k = 0; k < DO; k++) s += bvec[k] * c1W[k * DH + t];
    h1[t] = lrelu(s);
  }
  __syncthreads();
  if (t < 2){
    float s = c2b[t];
    for (int k = 0; k < DH; k++) s += h1[k] * c2W[k * 2 + t];
    out[t] = s;
  }
}

// ---------------------------------------------------------------------------
extern "C" void kernel_launch(void* const* d_in, const int* in_sizes, int n_in,
                              void* d_out, int out_size, void* d_ws, size_t ws_size,
                              hipStream_t stream)
{
  const float* x      = (const float*)d_in[0];
  const float* gumbel = (const float*)d_in[1];
  const float* enc_W  = (const float*)d_in[2];  const float* enc_b  = (const float*)d_in[3];
  const float* cl_W   = (const float*)d_in[4];  const float* cl_b   = (const float*)d_in[5];
  const float* Wh     = (const float*)d_in[6];  const float* bh     = (const float*)d_in[7];
  const float* Wt     = (const float*)d_in[8];  const float* bt     = (const float*)d_in[9];
  const float* sage_l_W = (const float*)d_in[10]; const float* sage_l_b = (const float*)d_in[11];
  const float* sage_r_W = (const float*)d_in[12];
  const float* ls_W = (const float*)d_in[13]; const float* ls_b = (const float*)d_in[14];
  const float* lb_W = (const float*)d_in[15]; const float* lb_b = (const float*)d_in[16];
  const float* gU_W = (const float*)d_in[17]; const float* gU_b = (const float*)d_in[18];
  const float* gV_W = (const float*)d_in[19]; const float* gV_b = (const float*)d_in[20];
  const float* gW_W = (const float*)d_in[21]; const float* gW_b = (const float*)d_in[22];
  const float* ln_w = (const float*)d_in[23]; const float* ln_b = (const float*)d_in[24];
  const float* feat_W = (const float*)d_in[25]; const float* feat_b = (const float*)d_in[26];
  const float* attA_W = (const float*)d_in[27]; const float* attB_W = (const float*)d_in[28];
  const float* attC_W = (const float*)d_in[29];
  const float* bl_W = (const float*)d_in[30]; const float* bl_b = (const float*)d_in[31];
  const float* ge1_W = (const float*)d_in[32]; const float* ge1_b = (const float*)d_in[33];
  const float* ge2_W = (const float*)d_in[34]; const float* ge2_b = (const float*)d_in[35];
  const float* fg_W = (const float*)d_in[36]; const float* fg_b = (const float*)d_in[37];
  const float* ft_W = (const float*)d_in[38]; const float* ft_b = (const float*)d_in[39];
  const float* c1_W = (const float*)d_in[40]; const float* c1_b = (const float*)d_in[41];
  const float* c2_W = (const float*)d_in[42]; const float* c2_b = (const float*)d_in[43];
  (void)in_sizes; (void)n_in; (void)out_size; (void)ws_size;

  // ---- workspace carve-up (~370 MB)
  char* wp = (char*)d_ws;
  auto alloc = [&](size_t bytes)->void*{ void* p = (void*)wp; wp += (bytes + 255) & ~(size_t)255; return p; };
  unsigned short* Hbf  = (unsigned short*)alloc((size_t)NSAMP * DH * 2);
  unsigned short* Pbf  = (unsigned short*)alloc((size_t)NSAMP * KC * 2);
  unsigned short* fbf  = (unsigned short*)alloc((size_t)NSAMP * DH * 2);
  unsigned short* sAbf = (unsigned short*)alloc((size_t)NSAMP * DO * 2);
  unsigned short* sBbf = (unsigned short*)alloc((size_t)NSAMP * DO * 2);
  unsigned short* encWt = (unsigned short*)alloc((size_t)DH * DIN * 2);  // [DH][DIN]
  unsigned short* clWt  = (unsigned short*)alloc((size_t)KC * DH * 2);   // [KC][DH]
  unsigned short* featWt= (unsigned short*)alloc((size_t)DH * DH * 2);   // [DH][DH]
  unsigned short* attAt = (unsigned short*)alloc((size_t)DO * DH * 2);   // [DO][DH]
  unsigned short* attBt = (unsigned short*)alloc((size_t)DO * DH * 2);
  float* scores = (float*)alloc((size_t)NSAMP * 4);
  float* parts  = (float*)alloc((size_t)KC * DH * 4);
  float* eh     = (float*)alloc((size_t)KC * DH * 4);
  float* et     = (float*)alloc((size_t)KC * DH * 4);
  float* LKK    = (float*)alloc((size_t)KC * KC * 4);
  float* wtop   = (float*)alloc((size_t)KC * TK * 4);
  int*   itop   = (int*)  alloc((size_t)KC * TK * 4);
  float* agg    = (float*)alloc((size_t)KC * DH * 4);
  float* cnt    = (float*)alloc((size_t)KC * 4);
  float* tmpA   = (float*)alloc((size_t)KC * DO * 4);
  float* Xg     = (float*)alloc((size_t)KC * DO * 4);
  float* summed = (float*)alloc((size_t)KC * DO * 4);
  float* XSs    = (float*)alloc((size_t)KC * DO * 4);
  float* XSm    = (float*)alloc((size_t)KC * DO * 4);
  float* sum_msg= (float*)alloc((size_t)KC * DO * 4);
  float* bi_msg = (float*)alloc((size_t)KC * DO * 4);
  float* t1     = (float*)alloc((size_t)KC * 64 * 4);
  float* gg     = (float*)alloc((size_t)KC * DO * 4);
  float* zn     = (float*)alloc((size_t)KC * DO * 4);
  float* t2     = (float*)alloc((size_t)KC * 64 * 4);
  float* gate   = (float*)alloc((size_t)KC * 4);
  float* bgnn   = (float*)alloc(DO * 4);
  float* pm     = (float*)alloc(512 * 4);
  float* gmax   = (float*)alloc(256);
  float* ps     = (float*)alloc(512 * 4);
  float* denom  = (float*)alloc(256);
  float* Af     = (float*)alloc(DH * 4);
  float* bbasic = (float*)alloc(DO * 4);

  auto cdiv = [](long a, long b){ return (int)((a + b - 1) / b); };

  // 0) transpose weights to bf16 [N][K] so B fragments load with unit k-stride
  k_transpose_bf16<<<cdiv((long)DIN*DH,256),256,0,stream>>>(enc_W, encWt, DIN, DH);
  k_transpose_bf16<<<cdiv((long)DH*KC,256),256,0,stream>>>(cl_W, clWt, DH, KC);
  k_transpose_bf16<<<cdiv((long)DH*DH,256),256,0,stream>>>(feat_W, featWt, DH, DH);
  k_transpose_bf16<<<cdiv((long)DH*DO,256),256,0,stream>>>(attA_W, attAt, DH, DO);
  k_transpose_bf16<<<cdiv((long)DH*DO,256),256,0,stream>>>(attB_W, attBt, DH, DO);

  // 1) H = lrelu(x @ enc_W + enc_b)  -> bf16              [100000 x 256]
  { long tiles = (long)(NSAMP/16) * (DH/16);
    k_gemm_wmma<1,0,1,1><<<cdiv(tiles,8),256,0,stream>>>(x, DIN, 1, encWt, DIN, 1, enc_b,
                                                         nullptr, Hbf, NSAMP, DIN, DH, 1, 1, 0); }
  // 2) P = softmax(2*(H@cl_W + cl_b + gumbel))  -> bf16   [100000 x 1024]
  k_cluster_softmax<<<NSAMP/16,256,0,stream>>>(Hbf, clWt, cl_b, gumbel, Pbf);
  // 3) parts = P^T @ H  (split-K=25, atomic accumulate)    [1024 x 256]
  k_zero<<<1024,256,0,stream>>>(parts, (long)KC*DH);
  { long tiles = (long)(KC/16) * (DH/16);     // 1024 tiles * 25 chunks = 25600 waves
    k_gemm_wmma<0,0,0,0><<<cdiv(tiles*25,8),256,0,stream>>>(Pbf, 1, KC, Hbf, 1, DH, nullptr,
                                                            parts, nullptr, KC, NSAMP, DH, 0, 25, 1); }
  // 4) edge logits + top-k + edge softmax
  k_gemm_f32<<<cdiv((long)KC*DH,256),256,0,stream>>>(parts,DH,1, Wh,DH,1, bh,nullptr,1.f, eh, KC,DH,DH,0);
  k_gemm_f32<<<cdiv((long)KC*DH,256),256,0,stream>>>(parts,DH,1, Wt,DH,1, bt,nullptr,1.f, et, KC,DH,DH,0);
  k_gemm_f32<<<cdiv((long)KC*KC,256),256,0,stream>>>(eh,DH,1, et,1,DH, nullptr,nullptr,0.0625f, LKK, KC,DH,KC,0);
  k_topk<<<KC/256,256,0,stream>>>(LKK, wtop, itop);
  // 5) SAGEConv (mean aggregation)
  k_zero<<<1024,256,0,stream>>>(agg, (long)KC*DH);
  k_zero<<<1,256,0,stream>>>(cnt, KC);
  k_sage_agg<<<KC*TK,DH,0,stream>>>(itop, parts, agg, cnt);
  k_divcnt<<<cdiv((long)KC*DH,256),256,0,stream>>>(agg, cnt, KC, DH);
  k_gemm_f32<<<cdiv((long)KC*DO,256),256,0,stream>>>(agg,DH,1, sage_l_W,DO,1, sage_l_b,nullptr,1.f, tmpA, KC,DH,DO,0);
  k_gemm_f32<<<cdiv((long)KC*DO,256),256,0,stream>>>(parts,DH,1, sage_r_W,DO,1, nullptr,tmpA,1.f, Xg, KC,DH,DO,1);
  // 6) weighted neighbor sum + gated combine + LayerNorm
  k_summed<<<cdiv((long)KC*DO,256),256,0,stream>>>(wtop, itop, Xg, summed);
  k_ewise2<<<cdiv((long)KC*DO,256),256,0,stream>>>(Xg, summed, XSs, XSm, (long)KC*DO);
  k_gemm_f32<<<cdiv((long)KC*DO,256),256,0,stream>>>(XSs,DO,1, ls_W,DO,1, ls_b,nullptr,1.f, sum_msg, KC,DO,DO,0);
  k_gemm_f32<<<cdiv((long)KC*DO,256),256,0,stream>>>(XSm,DO,1, lb_W,DO,1, lb_b,nullptr,1.f, bi_msg, KC,DO,DO,0);
  k_gemm_f32<<<cdiv((long)KC*64,256),256,0,stream>>>(Xg,DO,1, gU_W,64,1, gU_b,nullptr,1.f, t1, KC,DO,64,0);
  k_gemm_f32<<<cdiv((long)KC*64,256),256,0,stream>>>(summed,DO,1, gV_W,64,1, gV_b,t1,1.f, t1, KC,DO,64,0);
  k_gemm_f32<<<cdiv((long)KC*DO,256),256,0,stream>>>(t1,64,1, gW_W,DO,1, gW_b,nullptr,1.f, gg, KC,64,DO,2);
  k_out_ln<<<KC,DO,0,stream>>>(gg, sum_msg, bi_msg, Xg, ln_w, ln_b, zn);
  // 7) attentional aggregation of clusters -> b_gnn
  k_gemm_f32<<<cdiv((long)KC*64,256),256,0,stream>>>(zn,DO,1, ge1_W,64,1, ge1_b,nullptr,1.f, t2, KC,DO,64,1);
  k_gemm_f32<<<cdiv((long)KC,256),256,0,stream>>>(t2,64,1, ge2_W,1,1, ge2_b,nullptr,1.f, gate, KC,64,1,0);
  k_bgnn<<<1,256,0,stream>>>(gate, zn, bgnn);
  // 8) gated attention branch over samples
  { long tiles = (long)(NSAMP/16) * (DH/16);
    k_gemm_wmma<0,0,1,1><<<cdiv(tiles,8),256,0,stream>>>(Hbf, DH, 1, featWt, DH, 1, feat_b,
                                                         nullptr, fbf, NSAMP, DH, DH, 1, 1, 0); }
  { long tiles = (long)(NSAMP/16) * (DO/16);
    k_gemm_wmma<0,0,1,1><<<cdiv(tiles,8),256,0,stream>>>(fbf, DH, 1, attAt, DH, 1, nullptr,
                                                         nullptr, sAbf, NSAMP, DH, DO, 1, 1, 0); }
  { long tiles = (long)(NSAMP/16) * (DO/16);
    k_gemm_wmma<0,0,1,1><<<cdiv(tiles,8),256,0,stream>>>(fbf, DH, 1, attBt, DH, 1, nullptr,
                                                         nullptr, sBbf, NSAMP, DH, DO, 2, 1, 0); }
  k_scores<<<cdiv(NSAMP,256),256,0,stream>>>(sAbf, sBbf, attC_W, scores);
  int NB = (NSAMP + 255) / 256;
  k_red_part<<<NB,256,0,stream>>>(scores, NSAMP, nullptr, pm, 0);
  k_red_final<<<1,256,0,stream>>>(pm, NB, gmax, 0);
  k_red_part<<<NB,256,0,stream>>>(scores, NSAMP, gmax, ps, 1);
  k_red_final<<<1,256,0,stream>>>(ps, NB, denom, 1);
  k_att_pool<<<DH,256,0,stream>>>(scores, gmax, denom, fbf, Af);
  k_gemm_f32<<<1,256,0,stream>>>(Af,DH,1, bl_W,DO,1, bl_b,nullptr,1.f, bbasic, 1,DH,DO,1);
  // 9) fusion + classifier head -> [1,2] logits
  k_head<<<1,256,0,stream>>>(bgnn, bbasic, fg_W, fg_b, ft_W, ft_b,
                             c1_W, c1_b, c2_W, c2_b, (float*)d_out);
}
